// conv2d_fi_30889404793384
// MI455X (gfx1250) — matrix operand order repaired and checked by reading, not compiled, not run on hardware
//
#include <hip/hip_runtime.h>
#include <stdint.h>

// ---------------- problem constants ----------------
#define CO 64
#define CI 32
#define HH 56
#define WW 56
#define LL 3136           // 56*56
#define KK 288            // 32*3*3
#define KP 296            // padded LDS K-stride in halfs: 592B rows, 16B aligned,
                          // 148-word stride -> conflict-free 16-lane column reads
#define NT 196            // LL/16

#define THRESH_U24 134218u   // round(0.008 * 2^24): P(flip) = BER*BITS
#define SEED_P 0x9E3779B9u
#define SEED_C 0x85EBCA77u
#define SEED_Y 0xC2B2AE3Du

// ---------------- workspace layout (bytes) ----------------
#define OFF_SCAL  0         // uint32[16]: [0]=max|w| [1]=max|p| [2]=max|c| [3]=max|y| (float bits)
#define OFF_WCMAX 64        // float[288] per-k max|wq|
#define OFF_XCMAX 1216      // float[288] per-k max|xcol|
#define OFF_WCOL  4096      // _Float16[64*288] quantized weights
#define OFF_XCOL  40960     // _Float16[3136*288] im2col of x
// total ws use: 40960 + 1806336 = 1,847,296 bytes

typedef __attribute__((ext_vector_type(16))) _Float16     v16h;
typedef __attribute__((ext_vector_type(8)))  float        v8f;
typedef __attribute__((ext_vector_type(4)))  unsigned int v4u;

// ---------------- CDNA5 async global->LDS staging ----------------
#ifndef __has_builtin
#define __has_builtin(x) 0
#endif

#if __has_builtin(__builtin_amdgcn_global_load_async_to_lds_b32)
#define HAVE_ASYNC 1
#else
#define HAVE_ASYNC 0
#endif

#if HAVE_ASYNC && __has_builtin(__builtin_amdgcn_s_wait_asynccnt)
#define ASYNC_WAIT() __builtin_amdgcn_s_wait_asynccnt(0)
#elif HAVE_ASYNC
#define ASYNC_WAIT() asm volatile("s_wait_asynccnt 0" ::: "memory")
#else
#define ASYNC_WAIT() ((void)0)
#endif

__device__ __forceinline__ void lds_copy_b32(uint32_t* dst_lds, const uint32_t* src_g) {
#if HAVE_ASYNC
  __builtin_amdgcn_global_load_async_to_lds_b32(
      (__attribute__((address_space(1))) int*)src_g,
      (__attribute__((address_space(3))) int*)dst_lds, 0, 0);
#else
  *dst_lds = *src_g;
#endif
}

// ---------------- stateless per-element RNG + quant/flip ----------------
__device__ __forceinline__ uint32_t fi_rng(uint32_t idx, uint32_t seed) {
  uint32_t x = idx * 0x9E3779B9u + seed;
  x ^= x >> 16; x *= 0x7FEB352Du;
  x ^= x >> 15; x *= 0x846CA68Bu;
  x ^= x >> 16;
  return x;
}

// clip(round(v/s), -128, 127) then flip one random bit of the 2's-complement
// 8-bit code with prob BER*BITS; returns the (possibly flipped) signed code.
__device__ __forceinline__ int quant_flip(float v, float inv_s, uint32_t idx, uint32_t seed) {
  float r = rintf(v * inv_s);                 // round-half-even, as jnp.round
  r = fminf(fmaxf(r, -128.0f), 127.0f);
  int q = (int)r;
  uint32_t h = fi_rng(idx, seed);
  if ((h >> 8) < THRESH_U24) {
    int qf = (q & 255) ^ (1 << (h & 7));
    q = (qf & 0x80) ? (qf - 256) : qf;
  }
  return q;
}

// ---------------- K0: zero the atomic-max accumulators ----------------
__global__ void k0_init(void* ws) {
  uint32_t* scal  = (uint32_t*)((char*)ws + OFF_SCAL);
  uint32_t* xcmax = (uint32_t*)((char*)ws + OFF_XCMAX);
  int t = threadIdx.x;
  if (t < 16) scal[t] = 0u;
  int i = t - 16;
  if (i >= 0 && i < KK) xcmax[i] = 0u;
}

// ---------------- K1: weight absmax -> quantize -> per-k max ----------------
__global__ void k1_wprep(const float* __restrict__ w, void* ws) {
  __shared__ float red[256];
  int t = threadIdx.x;
  float m = 0.f;
  for (int i = t; i < CO * KK; i += 256) m = fmaxf(m, fabsf(w[i]));
  red[t] = m; __syncthreads();
  for (int s = 128; s > 0; s >>= 1) { if (t < s) red[t] = fmaxf(red[t], red[t + s]); __syncthreads(); }
  float sw  = red[0] / 127.0f + 1e-12f;
  float inv = 1.0f / sw;
  _Float16* wcol = (_Float16*)((char*)ws + OFF_WCOL);
  for (int i = t; i < CO * KK; i += 256) {
    float r = rintf(w[i] * inv);
    r = fminf(fmaxf(r, -128.f), 127.f);
    wcol[i] = (_Float16)(r * sw);           // weight.reshape(Co,-1) layout == linear layout
  }
  __syncthreads();
  uint32_t* wcmax = (uint32_t*)((char*)ws + OFF_WCMAX);
  uint32_t* scal  = (uint32_t*)((char*)ws + OFF_SCAL);
  if (t == 0) scal[0] = __float_as_uint(red[0]);
  for (int k = t; k < KK; k += 256) {
    float mm = 0.f;
    for (int co = 0; co < CO; ++co) mm = fmaxf(mm, fabsf((float)wcol[co * KK + k]));
    wcmax[k] = __float_as_uint(mm);
  }
}

// ---------------- K2: im2col (pad=1, stride=1) + per-k |x| max ----------------
__global__ void k2_im2col(const float* __restrict__ x, void* ws) {
  int k  = threadIdx.x;                 // 0..287, k = ci*9 + kh*3 + kw (C slowest)
  int lt = blockIdx.x;                  // L tile
  int ci = k / 9, rem = k % 9, kh = rem / 3, kw = rem % 3;
  _Float16* xcol  = (_Float16*)((char*)ws + OFF_XCOL);
  uint32_t* xcmax = (uint32_t*)((char*)ws + OFF_XCMAX);
  float m = 0.f;
  for (int i = 0; i < 16; ++i) {
    int l = lt * 16 + i;
    int r = l / WW, c = l % WW;
    int ih = r + kh - 1, iw = c + kw - 1;
    float v = 0.f;
    if ((unsigned)ih < (unsigned)HH && (unsigned)iw < (unsigned)WW)
      v = x[(ci * HH + ih) * WW + iw];
    _Float16 h = (_Float16)v;
    xcol[l * KK + k] = h;
    m = fmaxf(m, fabsf((float)h));
  }
  atomicMax(xcmax + k, __float_as_uint(m));
}

// ---------------- K3: s_p from per-k column maxima ----------------
__global__ void k3_pscale(void* ws) {
  __shared__ float red[256];
  int t = threadIdx.x;
  uint32_t* wcmax = (uint32_t*)((char*)ws + OFF_WCMAX);
  uint32_t* xcmax = (uint32_t*)((char*)ws + OFF_XCMAX);
  uint32_t* scal  = (uint32_t*)((char*)ws + OFF_SCAL);
  float m = 0.f;
  for (int k = t; k < KK; k += 256)
    m = fmaxf(m, __uint_as_float(xcmax[k]) * __uint_as_float(wcmax[k]));
  red[t] = m; __syncthreads();
  for (int s = 128; s > 0; s >>= 1) { if (t < s) red[t] = fmaxf(red[t], red[t + s]); __syncthreads(); }
  if (t == 0) scal[1] = __float_as_uint(red[0]);
}

// ---------------- K4: scan pass for s_c = max|cumsum(p_fi)| ----------------
__global__ void k4_cscale(void* ws) {
  __shared__ float red[256];
  int t   = threadIdx.x;
  int tid = blockIdx.x * 256 + t;       // tid = co*LL + l
  uint32_t* scal = (uint32_t*)((char*)ws + OFF_SCAL);
  const _Float16* wcol = (const _Float16*)((char*)ws + OFF_WCOL);
  const _Float16* xcol = (const _Float16*)((char*)ws + OFF_XCOL);
  float m = 0.f;
  if (tid < CO * LL) {
    int co = tid / LL, l = tid % LL;
    float sp = __uint_as_float(scal[1]) / 127.0f + 1e-12f;
    float inv_sp = 1.0f / sp;
    const _Float16* wr = wcol + co * KK;
    const _Float16* xr = xcol + l * KK;
    __builtin_prefetch(wr, 0, 0);       // global_prefetch_b8
    __builtin_prefetch(xr, 0, 0);
    uint32_t base = (uint32_t)tid * (uint32_t)KK;
    float clean = 0.f, corr = 0.f;
    for (int j = 0; j < KK; ++j) {
      float prod = (float)xr[j] * (float)wr[j];
      clean += prod;
      int pq = quant_flip(prod, inv_sp, base + (uint32_t)j, SEED_P);
      corr += (float)pq * sp - prod;    // quant residual + flip delta of p
      m = fmaxf(m, fabsf(clean + corr));
    }
  }
  red[t] = m; __syncthreads();
  for (int s = 128; s > 0; s >>= 1) { if (t < s) red[t] = fmaxf(red[t], red[t + s]); __syncthreads(); }
  if (t == 0) atomicMax(scal + 2, __float_as_uint(red[0]));
}

// ---------------- K5: WMMA GEMM + fused FI correction scan ----------------
// One block per 16-row L tile; 4 waves, wave w owns co-tile [16w, 16w+16).
__global__ void __launch_bounds__(128) k5_main(void* ws, float* __restrict__ ypre) {
  __shared__ __align__(16) _Float16 As[16 * KP];
  __shared__ __align__(16) _Float16 Bs[CO * KP];
  __shared__ float red[128];
  int t  = threadIdx.x;
  int l0 = blockIdx.x * 16;

  // ---- stage A (16x288) and B (64x288) into LDS, 296-half padded stride ----
  const uint32_t* xcolw = (const uint32_t*)((const char*)ws + OFF_XCOL);
  const uint32_t* wcolw = (const uint32_t*)((const char*)ws + OFF_WCOL);
  uint32_t* Asw = (uint32_t*)As;
  uint32_t* Bsw = (uint32_t*)Bs;
  for (int idx = t; idx < 16 * 144; idx += 128) {          // 144 words / row
    int r = idx / 144, wd = idx % 144;
    lds_copy_b32(Asw + r * (KP / 2) + wd, xcolw + (l0 + r) * 144 + wd);
  }
  for (int idx = t; idx < CO * 144; idx += 128) {
    int r = idx / 144, wd = idx % 144;
    lds_copy_b32(Bsw + r * (KP / 2) + wd, wcolw + r * 144 + wd);
  }
  ASYNC_WAIT();
  __syncthreads();

  int lane = t & 31;
  int ct   = t >> 5;                   // co tile 0..3
  int half = lane >> 4;                // 0/1
  int lm   = lane & 15;

  // ---- clean GEMM: 9 k-steps of v_wmma_f32_16x16x32_f16 ----
  const _Float16* ArowF = As + lm * KP;                 // A frag: M = lane&15
  const _Float16* BrowF = Bs + (ct * 16 + lm) * KP;     // B frag: N = lane&15
  int kbA = half * 8;                  // A halfs: K = kb..kb+7, kb+16..kb+23
  int kbB = half * 8;                  // B halfs: K = kb..kb+7, kb+16..kb+23
  union { v16h h; v4u q[2]; } af, bf;
  v8f acc = {};
  for (int ks = 0; ks < 9; ++ks) {
    int koff = ks * 32;
    af.q[0] = *(const v4u*)(ArowF + koff + kbA);
    af.q[1] = *(const v4u*)(ArowF + koff + kbA + 16);
    bf.q[0] = *(const v4u*)(BrowF + koff + kbB);
    bf.q[1] = *(const v4u*)(BrowF + koff + kbB + 16);
    acc = __builtin_amdgcn_wmma_f32_16x16x32_f16(
        false, af.h, false, bf.h, (short)0, acc, false, false);
  }

  // ---- FI correction scan: c_j = clean_j + corr_j; WMMA supplies clean_{K-1} ----
  uint32_t* scal = (uint32_t*)((char*)ws + OFF_SCAL);
  float sp = __uint_as_float(scal[1]) / 127.0f + 1e-12f;
  float sc = __uint_as_float(scal[2]) / 127.0f + 1e-12f;
  float inv_sp = 1.0f / sp, inv_sc = 1.0f / sc;
  int n = ct * 16 + lm;                // co
  const _Float16* br = Bs + n * KP;
  float ym = 0.f;
#pragma unroll
  for (int v = 0; v < 8; ++v) {
    int Mv = v + 8 * half;             // C/D layout: (VGPR v, lane) -> M,N
    int l  = l0 + Mv;
    const _Float16* ar = As + Mv * KP; // broadcast across the 16-lane half
    uint32_t base = (uint32_t)(n * LL + l) * (uint32_t)KK;
    float clean = 0.f, corr = 0.f, err = 0.f;
    for (int j = 0; j < KK - 1; ++j) {
      float prod = (float)ar[j] * (float)br[j];
      clean += prod;
      int pq = quant_flip(prod, inv_sp, base + (uint32_t)j, SEED_P);
      corr += (float)pq * sp - prod;
      float c = clean + corr;
      int cq = quant_flip(c, inv_sc, base + (uint32_t)j, SEED_C);
      if (j >= 1) err += (float)cq * sc - c;   // e_j for j in [1, K-2]
    }
    {  // j = K-1: substitute the WMMA clean sum for the scalar accumulator
      const int j = KK - 1;
      float prod = (float)ar[j] * (float)br[j];
      int pq = quant_flip(prod, inv_sp, base + (uint32_t)j, SEED_P);
      corr += (float)pq * sp - prod;
      float c = acc[v] + corr;                 // WMMA is load-bearing here
      int cq = quant_flip(c, inv_sc, base + (uint32_t)j, SEED_C);
      float y = (float)cq * sc + err;          // y_sum + c_error
      ypre[n * LL + l] = y;
      ym = fmaxf(ym, fabsf(y));
    }
  }
  red[t] = ym; __syncthreads();
  for (int s = 64; s > 0; s >>= 1) { if (t < s) red[t] = fmaxf(red[t], red[t + s]); __syncthreads(); }
  if (t == 0) atomicMax(scal + 3, __float_as_uint(red[0]));
}

// ---------------- K6: final output quant/flip (in place on d_out) ----------------
__global__ void k6_final(void* ws, float* __restrict__ out) {
  int tid = blockIdx.x * 256 + threadIdx.x;
  if (tid >= CO * LL) return;
  uint32_t* scal = (uint32_t*)((char*)ws + OFF_SCAL);
  float sy = __uint_as_float(scal[3]) / 127.0f + 1e-12f;
  float y  = out[tid];
  int q = quant_flip(y, 1.0f / sy, (uint32_t)tid, SEED_Y);
  out[tid] = (float)q * sy;
}

extern "C" void kernel_launch(void* const* d_in, const int* in_sizes, int n_in,
                              void* d_out, int out_size, void* d_ws, size_t ws_size,
                              hipStream_t stream) {
  const float* x = (const float*)d_in[0];   // (1,32,56,56)
  const float* w = (const float*)d_in[1];   // (64,32,3,3)
  float* out = (float*)d_out;               // (1,64,56,56)

  k0_init  <<<1,   320, 0, stream>>>(d_ws);
  k1_wprep <<<1,   256, 0, stream>>>(w, d_ws);
  k2_im2col<<<NT,  288, 0, stream>>>(x, d_ws);
  k3_pscale<<<1,   256, 0, stream>>>(d_ws);
  k4_cscale<<<784, 256, 0, stream>>>(d_ws);
  k5_main  <<<NT,  128, 0, stream>>>(d_ws, out);   // writes y_pre into d_out
  k6_final <<<784, 256, 0, stream>>>(d_ws, out);   // in-place final quant/flip
}
